// Head_4071628997029
// MI455X (gfx1250) — matrix operand order repaired and checked by reading, not compiled, not run on hardware
//
#include <hip/hip_runtime.h>

// ---------------------------------------------------------------------------
// Causal single-head attention for MI455X (gfx1250, wave32, WMMA bf16).
//   out = softmax(mask(QK^T * C^-0.5)) @ V,  Q/K/V = x @ W^T
// Working set (~77MB) is L2-resident (192MB L2) -> compute bound -> all
// matmuls on v_wmma_f32_16x16x32_bf16, fp32 accum, fp32 online softmax.
// K/V tiles staged to LDS with gfx1250 async-to-LDS loads (ASYNCcnt path).
// ---------------------------------------------------------------------------

typedef __attribute__((ext_vector_type(16))) __bf16 v16bf;
typedef __attribute__((ext_vector_type(8)))  float  v8f;
typedef int v4i __attribute__((vector_size(16)));

#define Tc 2048
#define Cc 1024
#define Hc 128
#define Bc 8

#if defined(__gfx1250__) && \
    __has_builtin(__builtin_amdgcn_global_load_async_to_lds_b128) && \
    __has_builtin(__builtin_amdgcn_s_wait_asynccnt)
#define HAVE_ASYNC_LDS 1
typedef __attribute__((address_space(1))) v4i as1_v4i;   // global b128
typedef __attribute__((address_space(3))) v4i as3_v4i;   // LDS b128
#endif

union FragBF {
    unsigned short u[16];
    uint4          q[2];
    v16bf          v;
};

// fp32 -> bf16 (RNE) via native cast so the compiler emits v_cvt ops,
// not integer round sequences.
__device__ inline unsigned short bf16u(float f) {
    union { __bf16 b; unsigned short u; } c;
    c.b = (__bf16)f;
    return c.u;
}

// ---------------------------------------------------------------------------
// Kernel 0: convert the three weight matrices to bf16 once.
// Wb layout: [3][Hc][Cc], mat order {q,k,v}. grid = (Hc*Cc/1024, 3) x 256.
// ---------------------------------------------------------------------------
__global__ __launch_bounds__(256)
void cvt_w_kernel(const float* __restrict__ Wq, const float* __restrict__ Wk,
                  const float* __restrict__ Wv, unsigned short* __restrict__ Wb)
{
    const int tid = blockIdx.x * 256 + threadIdx.x;     // one thread per 4 elems
    const int mat = blockIdx.y;
    const float* src = (mat == 0) ? Wq : (mat == 1) ? Wk : Wv;
    float4 f = ((const float4*)src)[tid];
    union { unsigned short u[4]; uint2 v; } o;
    o.u[0] = bf16u(f.x); o.u[1] = bf16u(f.y);
    o.u[2] = bf16u(f.z); o.u[3] = bf16u(f.w);
    ((uint2*)(Wb + (size_t)mat * Hc * Cc))[tid] = o.v;
}

// ---------------------------------------------------------------------------
// Kernel 1: Q/K/V projection. One wave owns a 16-row tile and 12 of the 24
// output column tiles (3 mats x 8 head tiles), so each x chunk is converted
// to bf16 once and reused by 12 WMMAs. grid = (B*T/16, 2).
// ---------------------------------------------------------------------------
__global__ __launch_bounds__(32)
void proj_qkv_kernel(const float* __restrict__ x,
                     const unsigned short* __restrict__ Wb,
                     unsigned short* __restrict__ qb,
                     unsigned short* __restrict__ kb,
                     unsigned short* __restrict__ vb)
{
    const int lane = threadIdx.x;
    const int m0   = blockIdx.x * 16;
    const int g    = blockIdx.y;            // col-tile group: [g*12, g*12+12)
    const int am   = lane & 15;             // A row / B column
    const int half = lane >> 4;
    const int ak   = half * 16;             // unused here (B K base is half*8)

    const float* xrow = x + (size_t)(m0 + am) * Cc;

    v8f acc[12];
#pragma unroll
    for (int t = 0; t < 12; ++t) acc[t] = (v8f)(0.0f);

    for (int cb = 0; cb < Cc; cb += 32) {
        // A fragment (ISA 7.12.2 16-bit A layout): convert once per chunk
        FragBF a;
#pragma unroll
        for (int j = 0; j < 8; ++j) {
            a.u[j]     = bf16u(xrow[cb + half * 8 + j]);
            a.u[8 + j] = bf16u(xrow[cb + 16 + half * 8 + j]);
        }
#pragma unroll
        for (int t = 0; t < 12; ++t) {
            const int ct = g * 12 + t;      // global col tile 0..23
            // B frag: col = lane&15, K = half*8 + j and 16 + half*8 + j -> 2 x b128 loads
            const unsigned short* brow =
                Wb + (size_t)(ct * 16 + am) * Cc + cb + half * 8;
            FragBF bfr;
            bfr.q[0] = *(const uint4*)(brow);
            bfr.q[1] = *(const uint4*)(brow + 16);
            acc[t] = __builtin_amdgcn_wmma_f32_16x16x32_bf16(
                false, a.v, false, bfr.v, (short)0, acc[t], false, false);
        }
    }

    // C/D layout: col = lane&15, row M = half*8 + r
#pragma unroll
    for (int t = 0; t < 12; ++t) {
        const int ct  = g * 12 + t;
        const int mat = ct >> 3, nt = ct & 7;
        unsigned short* dst = (mat == 0) ? qb : (mat == 1) ? kb : vb;
#pragma unroll
        for (int r = 0; r < 8; ++r)
            dst[(size_t)(m0 + half * 8 + r) * Hc + nt * 16 + am] =
                bf16u(acc[t][r]);
    }
}

// ---------------------------------------------------------------------------
// Kernel 2: flash attention, one wave per 16 query rows. Key blocks of 32
// staged in LDS via async-to-LDS loads; fp32 online softmax; O kept in the
// WMMA C operand. grid = (T/16, B).
// ---------------------------------------------------------------------------
__global__ __launch_bounds__(32)
void flash_attn_kernel(const unsigned short* __restrict__ qb,
                       const unsigned short* __restrict__ kb,
                       const unsigned short* __restrict__ vb,
                       float* __restrict__ out)
{
    __shared__ __align__(16) unsigned short Kt[32 * Hc];   // 8 KB
    __shared__ __align__(16) unsigned short Vt[32 * Hc];   // 8 KB
    __shared__ __align__(16) unsigned short Pt[16 * 32];   // 1 KB

    const int lane = threadIdx.x;
    const int q0   = blockIdx.x * 16;
    const int b    = blockIdx.y;
    const size_t base = (size_t)b * Tc * Hc;

    const int am   = lane & 15;
    const int half = lane >> 4;
    const int ak   = half * 16;

    // Q fragments (4 K-chunks over H=128), b128 loads
    FragBF qf[4];
    {
        const unsigned short* qrow = qb + base + (size_t)(q0 + am) * Hc;
#pragma unroll
        for (int hc = 0; hc < 4; ++hc) {
            qf[hc].q[0] = *(const uint4*)(qrow + hc * 32 + half * 8);
            qf[hc].q[1] = *(const uint4*)(qrow + hc * 32 + 16 + half * 8);
        }
    }

    v8f o[8] = {};                          // 8 head tiles of 16x16 fp32
    float m_run[8], l_run[8];
#pragma unroll
    for (int r = 0; r < 8; ++r) { m_run[r] = -3.0e38f; l_run[r] = 0.0f; }

    const float scale = 0.03125f;           // C^-0.5 = 1/sqrt(1024)

    const int nblocks = (q0 + 47) / 32;     // key blocks covering keys <= q0+15
    for (int jb = 0; jb < nblocks; ++jb) {
        const int j0 = jb * 32;

        // ---- stage K,V rows [j0, j0+32) into LDS, one row per lane ----
        {
            const unsigned short* kg = kb + base + (size_t)(j0 + lane) * Hc;
            const unsigned short* vg = vb + base + (size_t)(j0 + lane) * Hc;
            unsigned short* kl = &Kt[lane * Hc];
            unsigned short* vl = &Vt[lane * Hc];
#ifdef HAVE_ASYNC_LDS
#pragma unroll
            for (int i = 0; i < Hc / 8; ++i) {
                __builtin_amdgcn_global_load_async_to_lds_b128(
                    (as1_v4i*)(kg + i * 8), (as3_v4i*)(kl + i * 8), 0, 0);
                __builtin_amdgcn_global_load_async_to_lds_b128(
                    (as1_v4i*)(vg + i * 8), (as3_v4i*)(vl + i * 8), 0, 0);
            }
            __builtin_amdgcn_s_wait_asynccnt(0);
#else
#pragma unroll
            for (int i = 0; i < Hc / 8; ++i) {
                ((uint4*)kl)[i] = ((const uint4*)kg)[i];
                ((uint4*)vl)[i] = ((const uint4*)vg)[i];
            }
            __syncthreads();
#endif
        }

        // ---- S = Q @ K^T : two 16x16 fp32 tiles, K=128 contraction ----
        v8f s_acc[2];
#pragma unroll
        for (int nh = 0; nh < 2; ++nh) {
            s_acc[nh] = (v8f)(0.0f);
#pragma unroll
            for (int hc = 0; hc < 4; ++hc) {
                FragBF kf;  // B frag: col = key (nh*16+am), K = h = half*8+j, 16+half*8+j
                const unsigned short* kp =
                    &Kt[(nh * 16 + am) * Hc + hc * 32 + half * 8];
                kf.q[0] = *(const uint4*)(kp);
                kf.q[1] = *(const uint4*)(kp + 16);
                s_acc[nh] = __builtin_amdgcn_wmma_f32_16x16x32_bf16(
                    false, qf[hc].v, false, kf.v, (short)0, s_acc[nh],
                    false, false);
            }
        }

        // ---- fp32 online softmax. Row M = half*8 + r lives in one 16-lane
        //      half; xor masks {8,4,2,1} reduce the 16 columns of a row ----
        float mnew[8], alpha[8], p0[8], p1[8];
#pragma unroll
        for (int r = 0; r < 8; ++r) {
            const int trow = q0 + half * 8 + r;
            float v0 = s_acc[0][r] * scale;
            float v1 = s_acc[1][r] * scale;
            if (j0 + am > trow)      v0 = -3.0e38f;   // causal mask
            if (j0 + 16 + am > trow) v1 = -3.0e38f;
            p0[r] = v0; p1[r] = v1;
            float mv = fmaxf(v0, v1);
#pragma unroll
            for (int off = 8; off >= 1; off >>= 1)
                mv = fmaxf(mv, __shfl_xor(mv, off, 32));
            mnew[r]  = fmaxf(m_run[r], mv);
            alpha[r] = __expf(m_run[r] - mnew[r]);
        }
#pragma unroll
        for (int r = 0; r < 8; ++r) {
            float e0 = __expf(p0[r] - mnew[r]);
            float e1 = __expf(p1[r] - mnew[r]);
            p0[r] = e0; p1[r] = e1;
            float sv = e0 + e1;
#pragma unroll
            for (int off = 8; off >= 1; off >>= 1)
                sv += __shfl_xor(sv, off, 32);
            l_run[r] = l_run[r] * alpha[r] + sv;
            m_run[r] = mnew[r];
        }

        // ---- P: C/D layout -> A layout via LDS bounce (bf16) ----
#pragma unroll
        for (int r = 0; r < 8; ++r) {
            Pt[(half * 8 + r) * 32 + am]      = bf16u(p0[r]);
            Pt[(half * 8 + r) * 32 + 16 + am] = bf16u(p1[r]);
        }
#ifndef HAVE_ASYNC_LDS
        __syncthreads();
#endif
        FragBF pf;
        pf.q[0] = *(const uint4*)(&Pt[am * 32 + half * 16]);
        pf.q[1] = *(const uint4*)(&Pt[am * 32 + half * 16 + 8]);

        // ---- O = alpha*O + P @ V (accumulate through WMMA C operand) ----
#pragma unroll
        for (int t = 0; t < 8; ++t) {
            v8f oc;
#pragma unroll
            for (int r = 0; r < 8; ++r) oc[r] = o[t][r] * alpha[r];
            FragBF vf;  // B frag: col = head col (am), K = s = ak+j (strided)
#pragma unroll
            for (int j = 0; j < 16; ++j)
                vf.u[j] = Vt[(ak + j) * Hc + t * 16 + am];
            o[t] = __builtin_amdgcn_wmma_f32_16x16x32_bf16(
                false, pf.v, false, vf.v, (short)0, oc, false, false);
        }
#ifndef HAVE_ASYNC_LDS
        __syncthreads();   // protect Kt/Vt before next block's staging
#endif
    }

    // ---- normalize and write fp32 output ----
#pragma unroll
    for (int t = 0; t < 8; ++t) {
#pragma unroll
        for (int r = 0; r < 8; ++r) {
            const int row = q0 + half * 8 + r;
            out[base + (size_t)row * Hc + t * 16 + am] = o[t][r] / l_run[r];
        }
    }
}

// ---------------------------------------------------------------------------
// Launch: inputs = {x, Wk, Wq, Wv} fp32.
// Scratch: Wb (768KB bf16) + Q/K/V (12MB bf16) ~= 12.8MB of d_ws.
// ---------------------------------------------------------------------------
extern "C" void kernel_launch(void* const* d_in, const int* in_sizes, int n_in,
                              void* d_out, int out_size, void* d_ws, size_t ws_size,
                              hipStream_t stream) {
    const float* x  = (const float*)d_in[0];
    const float* Wk = (const float*)d_in[1];
    const float* Wq = (const float*)d_in[2];
    const float* Wv = (const float*)d_in[3];
    float* out = (float*)d_out;

    const size_t N = (size_t)Bc * Tc * Hc;            // elements per Q/K/V
    unsigned short* Wb = (unsigned short*)d_ws;       // [3][Hc][Cc] bf16
    unsigned short* qb = Wb + (size_t)3 * Hc * Cc;
    unsigned short* kb = qb + N;
    unsigned short* vb = kb + N;

    dim3 g0((Hc * Cc) / 1024, 3);                     // weight bf16 convert
    cvt_w_kernel<<<g0, 256, 0, stream>>>(Wq, Wk, Wv, Wb);

    dim3 g1((Bc * Tc) / 16, 2);                       // 1024 row tiles x 2 groups
    proj_qkv_kernel<<<g1, 32, 0, stream>>>(x, Wb, qb, kb, vb);

    dim3 g2(Tc / 16, Bc);                             // 128 query tiles x 8 batches
    flash_attn_kernel<<<g2, 32, 0, stream>>>(qb, kb, vb, out);
}
